// SliceAttention_23931557773503
// MI455X (gfx1250) — matrix operand order repaired and checked by reading, not compiled, not run on hardware
//
#include <hip/hip_runtime.h>
#include <hip/hip_bf16.h>
#include <math.h>

// Problem constants (B=4, L=128, N=512)
#define BB 4
#define LL 128
#define NN 512
#define MM (BB * LL * LL) /* 65536 rows for the projection GEMM */
static constexpr float QK_SCALE = 0.13245323570650439f; // 1/sqrt(57)

typedef __bf16 bf16;
typedef bf16  bf16x4  __attribute__((ext_vector_type(4)));
typedef bf16  bf16x8  __attribute__((ext_vector_type(8)));
typedef bf16  bf16x16 __attribute__((ext_vector_type(16)));
typedef float floatx8 __attribute__((ext_vector_type(8)));

union FragU { bf16x16 v; bf16x8 h[2]; };

__device__ inline floatx8 zero8() {
  floatx8 z;
#pragma unroll
  for (int e = 0; e < 8; ++e) z[e] = 0.0f;
  return z;
}

// Async global->LDS 16-byte copy (CDNA5 TDM-class async path, ASYNCcnt-tracked).
// GV mode: vdst = LDS byte address VGPR, vaddr = 64-bit global address pair.
__device__ inline void async_copy_b128(void* lds_ptr, const void* gptr) {
  const uint32_t lds_addr = (uint32_t)(uintptr_t)lds_ptr;
  asm volatile("global_load_async_to_lds_b128 %0, %1, off"
               :
               : "v"(lds_addr), "v"((uint64_t)(uintptr_t)gptr)
               : "memory");
}
__device__ inline void wait_asynccnt0() {
  asm volatile("s_wait_asynccnt 0" ::: "memory");
}

// Padded LDS row stride (in bf16 elements): 40 bf16 = 20 dwords per row.
// Lane rows r*20 dwords apart -> 16 consecutive rows map to 16 distinct banks.
#define LDS_STRIDE 40

// ---------------------------------------------------------------------------
// float32 -> bf16 elementwise convert (n must be a multiple of 4)
// ---------------------------------------------------------------------------
__global__ void cvt_f32_to_bf16(const float* __restrict__ in,
                                bf16* __restrict__ out, int n) {
  int i = (blockIdx.x * blockDim.x + threadIdx.x) * 4;
  if (i + 3 < n) {
    float4 v = *(const float4*)(in + i);
    bf16x4 o;
    o[0] = (bf16)v.x; o[1] = (bf16)v.y; o[2] = (bf16)v.z; o[3] = (bf16)v.w;
    *(bf16x4*)(out + i) = o;
  }
}

// ---------------------------------------------------------------------------
// rowsum[b,i,l] = sum_n outer[b,i,l,n]  (one wave32 per 512-float row)
// ---------------------------------------------------------------------------
__global__ void rowsum_kernel(const float* __restrict__ outer,
                              float* __restrict__ rowsum) {
  const int row  = blockIdx.x * 8 + (threadIdx.x >> 5); // 65536 rows
  const int lane = threadIdx.x & 31;
  const float4* p = (const float4*)(outer + (size_t)row * NN);
  float s = 0.0f;
#pragma unroll
  for (int t = 0; t < 4; ++t) {
    float4 v = p[lane + t * 32];
    s += v.x + v.y + v.z + v.w;
  }
#pragma unroll
  for (int off = 16; off > 0; off >>= 1) s += __shfl_xor(s, off, 32);
  if (lane == 0) rowsum[row] = s;
}

// ---------------------------------------------------------------------------
// Projection GEMM: K/Q = sigmoid(X @ W^T + b), X:[M,512] bf16, W:[512,512] bf16
// Two blocks per 16-row m-tile (nhalf = 0/1); wave w covers n-tiles
// nhalf*16 + 2w + {0,1} for BOTH weights -> 4 accumulators/wave (32 VGPRs).
// Per k-step: async-stage A tile to LDS, batch-load 4 B fragments, 4 WMMAs.
// A-frag (16x32 bf16): lane m=lane&15, half=lane>>4 -> K chunk half*16 (contiguous, same as B)
// B-frag (32x16 bf16): lane n=lane&15, K chunk half*16 (W rows are K-contiguous)
// ---------------------------------------------------------------------------
__global__ void proj_gemm_kernel(const bf16* __restrict__ X,
                                 const bf16* __restrict__ Ws,
                                 const bf16* __restrict__ Wq,
                                 const float* __restrict__ bs,
                                 const float* __restrict__ bq,
                                 bf16* __restrict__ Kout,
                                 bf16* __restrict__ Qout) {
  __shared__ bf16 atile[16 * LDS_STRIDE];   // 1.25 KB
  const int tid    = threadIdx.x;
  const int lane   = tid & 31;
  const int wave   = tid >> 5;              // 0..7
  const int mtile  = blockIdx.x >> 1;       // 4096 m-tiles
  const int nhalf  = blockIdx.x & 1;        // which 16 n-tiles
  const int m0     = mtile * 16;
  const int half   = lane >> 4;
  const int nt0    = nhalf * 16 + wave * 2; // first n-tile of this wave

  floatx8 accS[2], accQ[2];
#pragma unroll
  for (int t = 0; t < 2; ++t) { accS[t] = zero8(); accQ[t] = zero8(); }

#pragma unroll 1
  for (int k0 = 0; k0 < NN; k0 += 32) {
    __syncthreads(); // protect previous iteration's LDS reads
    if (tid < 64) {  // 64 lanes x 16 B = 16 rows x 32 cols (async to LDS)
      const int r = tid >> 2;
      const int c = (tid & 3) * 8;
      async_copy_b128(&atile[r * LDS_STRIDE + c],
                      X + (size_t)(m0 + r) * NN + k0 + c);
    }
    wait_asynccnt0();
    __syncthreads();

    // ---- load phase: A from LDS + all 4 B fragments from global ----
    FragU a;
    const bf16* ap = &atile[(lane & 15) * LDS_STRIDE + half * 16];
    a.h[0] = *(const bf16x8*)ap;
    a.h[1] = *(const bf16x8*)(ap + 8);

    FragU bS[2], bQ[2];
#pragma unroll
    for (int t = 0; t < 2; ++t) {
      const int n = (nt0 + t) * 16 + (lane & 15);
      const bf16* bp = Ws + (size_t)n * NN + k0 + half * 16;
      bS[t].h[0] = *(const bf16x8*)bp;
      bS[t].h[1] = *(const bf16x8*)(bp + 8);
      const bf16* cp = Wq + (size_t)n * NN + k0 + half * 16;
      bQ[t].h[0] = *(const bf16x8*)cp;
      bQ[t].h[1] = *(const bf16x8*)(cp + 8);
      if (k0 + 32 < NN) __builtin_prefetch(bp + 32, 0, 1);
    }

    // ---- compute phase: 4 WMMAs back-to-back ----
#pragma unroll
    for (int t = 0; t < 2; ++t) {
      accS[t] = __builtin_amdgcn_wmma_f32_16x16x32_bf16(
          false, a.v, false, bS[t].v, (short)0, accS[t], false, false);
      accQ[t] = __builtin_amdgcn_wmma_f32_16x16x32_bf16(
          false, a.v, false, bQ[t].v, (short)0, accQ[t], false, false);
    }
  }

  // Epilogue: bias + sigmoid, store bf16 row-major [m][n].
  // C layout: VGPR r, lane l -> M = r + 8*(l>>4), N = l&15.
#pragma unroll
  for (int t = 0; t < 2; ++t) {
    const int n = (nt0 + t) * 16 + (lane & 15);
    const float biasS = bs[n];
    const float biasQ = bq[n];
#pragma unroll
    for (int r = 0; r < 8; ++r) {
      const int m = m0 + r + 8 * half;
      const float xs = accS[t][r] + biasS;
      const float xq = accQ[t][r] + biasQ;
      Kout[(size_t)m * NN + n] = (bf16)(1.0f / (1.0f + __expf(-xs)));
      Qout[(size_t)m * NN + n] = (bf16)(1.0f / (1.0f + __expf(-xq)));
    }
  }
}

// ---------------------------------------------------------------------------
// s[b,i,j,l] = SCALE * sum_n q[b,i,j,n]*k[b,i,l,n] + mask[b,i]
// Two blocks per (b,i) (lhalf = 0/1): block covers l-tiles lhalf*4..+3.
// Wave = j-tile; 4 accumulators/wave. 64x32 k-tile async-staged in LDS per
// k-step (one async b128 per thread), then 4 batched B frags + 4 WMMAs.
// ---------------------------------------------------------------------------
__global__ void qk_gemm_kernel(const bf16* __restrict__ Q,
                               const bf16* __restrict__ K,
                               const float* __restrict__ mask,
                               float* __restrict__ S) {
  __shared__ bf16 ktile[64 * LDS_STRIDE];   // 5 KB
  const int tid   = threadIdx.x;
  const int lane  = tid & 31;
  const int jt    = tid >> 5;        // j-tile 0..7
  const int bi    = blockIdx.x >> 1; // b*L + i
  const int lhalf = blockIdx.x & 1;  // which 4 l-tiles
  const int half  = lane >> 4;
  const bf16* qb = Q + (size_t)bi * LL * NN;
  const bf16* kb = K + (size_t)bi * LL * NN + (size_t)(lhalf * 64) * NN;

  floatx8 acc[4];
#pragma unroll
  for (int lt = 0; lt < 4; ++lt) acc[lt] = zero8();

  const bf16* arow = qb + (size_t)(jt * 16 + (lane & 15)) * NN;
#pragma unroll 1
  for (int k0 = 0; k0 < NN; k0 += 32) {
    __syncthreads(); // protect previous iteration's LDS reads
    {
      // 256 threads x 16 B = 64 rows x 32 cols (async to LDS)
      const int r = tid >> 2;          // 0..63
      const int c = (tid & 3) * 8;
      async_copy_b128(&ktile[r * LDS_STRIDE + c],
                      kb + (size_t)r * NN + k0 + c);
    }
    wait_asynccnt0();
    __syncthreads();

    // ---- load phase: A from global + all 4 B fragments from LDS ----
    FragU a;
    const bf16* ap = arow + k0 + half * 16;
    a.h[0] = *(const bf16x8*)ap;
    a.h[1] = *(const bf16x8*)(ap + 8);
    if (k0 + 32 < NN) __builtin_prefetch(ap + 32, 0, 1);

    FragU bfr[4];
#pragma unroll
    for (int lt = 0; lt < 4; ++lt) {
      const int lr = lt * 16 + (lane & 15); // local row within the 64-row tile
      const bf16* bp = &ktile[lr * LDS_STRIDE + half * 16];
      bfr[lt].h[0] = *(const bf16x8*)bp;
      bfr[lt].h[1] = *(const bf16x8*)(bp + 8);
    }

    // ---- compute phase: 4 WMMAs back-to-back ----
#pragma unroll
    for (int lt = 0; lt < 4; ++lt) {
      acc[lt] = __builtin_amdgcn_wmma_f32_16x16x32_bf16(
          false, a.v, false, bfr[lt].v, (short)0, acc[lt], false, false);
    }
  }

  const float mval = mask[bi]; // mask broadcast value depends only on (b,i)
  float* sb = S + (size_t)bi * LL * LL;
#pragma unroll
  for (int lt = 0; lt < 4; ++lt) {
    const int l = lhalf * 64 + lt * 16 + (lane & 15);
#pragma unroll
    for (int r = 0; r < 8; ++r) {
      const int j = jt * 16 + r + 8 * half;
      sb[(size_t)j * LL + l] = acc[lt][r] * QK_SCALE + mval;
    }
  }
}

// ---------------------------------------------------------------------------
// Online softmax stats over axis i: for each (b,j,l) compute (max_i, 1/sumexp_i)
// Block = (b,j) [512 blocks], thread = l [128 threads]; stride L*L over i.
// ---------------------------------------------------------------------------
__global__ void softmax_stats_kernel(const float* __restrict__ S,
                                     float2* __restrict__ stats) {
  const int bj = blockIdx.x;       // b*L + j
  const int b  = bj >> 7;
  const int j  = bj & 127;
  const int l  = threadIdx.x;
  const float* sp = S + (size_t)b * LL * LL * LL + (size_t)j * LL + l;
  float m = -INFINITY, z = 0.0f;
#pragma unroll 1
  for (int i = 0; i < LL; ++i) {
    const float v  = sp[(size_t)i * LL * LL];
    const float nm = fmaxf(m, v);
    z = z * __expf(m - nm) + __expf(v - nm);
    m = nm;
  }
  float2 r; r.x = m; r.y = 1.0f / z;
  stats[(size_t)bj * LL + l] = r;
}

// ---------------------------------------------------------------------------
// out[b,i,j] = sum_l softmax_i(s)[b,i,j,l] * rowsum[b,i,l]
// (n-sum commuted through the second einsum: attn never materialized)
// ---------------------------------------------------------------------------
__global__ void out_final_kernel(const float* __restrict__ S,
                                 const float2* __restrict__ stats,
                                 const float* __restrict__ rowsum,
                                 float* __restrict__ out) {
  const int idx = blockIdx.x * blockDim.x + threadIdx.x; // (b*L+i)*L + j
  const int b = idx >> 14;
  const int i = (idx >> 7) & 127;
  const int j = idx & 127;
  const float*  srow = S + (size_t)idx * LL;
  const float2* st   = stats + ((size_t)(b * LL + j)) * LL;
  const float*  rs   = rowsum + ((size_t)(b * LL + i)) * LL;
  float acc = 0.0f;
#pragma unroll 4
  for (int l = 0; l < LL; ++l) {
    const float2 mz = st[l];
    acc += __expf(srow[l] - mz.x) * mz.y * rs[l];
  }
  out[idx] = acc;
}

// ---------------------------------------------------------------------------
// Host-side launch. Workspace layout (bytes):
//   outer_bf : [M,512] bf16           @ 0          (67,108,864)
//   Ws_bf    : [512,512] bf16         @ 67,108,864 (  524,288)
//   Wq_bf    : [512,512] bf16         @ 67,633,152 (  524,288)
//   qbf      : [M,512] bf16           @ 68,157,440 (67,108,864)
//   kbf      : [M,512] bf16           @ 135,266,304(67,108,864)
//   s        : [B,L,L,L] f32          @ 202,375,168(33,554,432)
//   stats    : [B,L,L] float2         @ 235,929,600(  524,288)
//   rowsum   : [B,L,L] f32            @ 236,453,888(  262,144)
// ---------------------------------------------------------------------------
extern "C" void kernel_launch(void* const* d_in, const int* in_sizes, int n_in,
                              void* d_out, int out_size, void* d_ws, size_t ws_size,
                              hipStream_t stream) {
  const float* outer = (const float*)d_in[0]; // [B,L,L,N] f32
  const float* mask  = (const float*)d_in[1]; // [B,1,1,L] f32
  const float* Wsl   = (const float*)d_in[2]; // [N,N]
  const float* bsl   = (const float*)d_in[3]; // [N]
  const float* Wqr   = (const float*)d_in[4]; // [N,N]
  const float* bqr   = (const float*)d_in[5]; // [N]
  float* out = (float*)d_out;                 // [B,L,L] f32

  char* ws = (char*)d_ws;
  bf16*   outer_bf = (bf16*)(ws + 0);
  bf16*   Ws_bf    = (bf16*)(ws + 67108864ull);
  bf16*   Wq_bf    = (bf16*)(ws + 67633152ull);
  bf16*   qbf      = (bf16*)(ws + 68157440ull);
  bf16*   kbf      = (bf16*)(ws + 135266304ull);
  float*  sbuf     = (float*)(ws + 202375168ull);
  float2* stats    = (float2*)(ws + 235929600ull);
  float*  rowsum   = (float*)(ws + 236453888ull);

  // 1) precision-convert inputs to bf16
  cvt_f32_to_bf16<<<(MM * NN / 4 + 255) / 256, 256, 0, stream>>>(outer, outer_bf, MM * NN);
  cvt_f32_to_bf16<<<(NN * NN / 4 + 255) / 256, 256, 0, stream>>>(Wsl, Ws_bf, NN * NN);
  cvt_f32_to_bf16<<<(NN * NN / 4 + 255) / 256, 256, 0, stream>>>(Wqr, Wq_bf, NN * NN);

  // 2) rowsum over n (enables eliminating the final GEMM)
  rowsum_kernel<<<MM / 8, 256, 0, stream>>>(outer, rowsum);

  // 3) fused dual projection GEMM + bias + sigmoid (WMMA bf16, async LDS A-tile)
  proj_gemm_kernel<<<(MM / 16) * 2, 256, 0, stream>>>(outer_bf, Ws_bf, Wq_bf, bsl, bqr, kbf, qbf);

  // 4) batched q.k^T GEMM + scale + mask (WMMA bf16, async LDS k-tile)
  qk_gemm_kernel<<<BB * LL * 2, 256, 0, stream>>>(qbf, kbf, mask, sbuf);

  // 5) softmax over axis i: online max/sumexp stats per (b,j,l)
  softmax_stats_kernel<<<BB * LL, LL, 0, stream>>>(sbuf, stats);

  // 6) normalize and contract with rowsum -> final [B,L,L]
  out_final_kernel<<<MM / 256, 256, 0, stream>>>(sbuf, stats, rowsum, out);

  (void)in_sizes; (void)n_in; (void)out_size; (void)ws_size;
}
